// BiMultiHeadAttention_16071767622214
// MI455X (gfx1250) — matrix operand order repaired and checked by reading, not compiled, not run on hardware
//
#include <hip/hip_runtime.h>
#include <stdint.h>

typedef unsigned short u16;
typedef unsigned int   u32;
typedef __attribute__((ext_vector_type(16))) __bf16 v16bf;
typedef __attribute__((ext_vector_type(4)))  __bf16 v4bf;
typedef __attribute__((ext_vector_type(8)))  float  v8f;
typedef __attribute__((ext_vector_type(4)))  float  v4f;
typedef __attribute__((ext_vector_type(4)))  u32    u32x4;

union Frag {
  v16bf v;
  u32x4 q[2];
  v4bf  b4[4];
  u16   h[16];
};

__device__ __forceinline__ u16 f2bf(float f) {
  return __builtin_bit_cast(u16, (__bf16)f);   // hardware cvt, RNE
}

__device__ __forceinline__ v8f wmma_bf16(Frag a, Frag b, v8f c) {
  return __builtin_amdgcn_wmma_f32_16x16x32_bf16(false, a.v, false, b.v,
                                                 (short)0, c, false, false);
}

// ---------------------------------------------------------------------------
// Generic WMMA GEMM: C[M,N] = alpha * (A[M,K] @ WT[N,K]^T + bias[N])
// Compile-time: AF32 (A dtype), MODE:
//   0: f32 row-major [M][N]
//   1: bf16 head layout  [B,H,rows,32]   (n = h*32+d, m = b*rows+t)
//   2: bf16 head layout transposed [B,H,32,rows]
// One wave per 16x16 output tile; 8 waves / block.
// ---------------------------------------------------------------------------
template<bool AF32, int MODE>
__global__ __launch_bounds__(256)
void gemm_wmma(const void* __restrict__ Ap, const u16* __restrict__ WT,
               const float* __restrict__ bias, float alpha,
               void* __restrict__ outp, int M, int N, int K, int rows)
{
  const int lane = threadIdx.x & 31;
  const int wave = threadIdx.x >> 5;
  const int half = lane >> 4;
  const int l15  = lane & 15;

  const int  ntiles = N >> 4;
  const long tile   = (long)blockIdx.x * 8 + wave;
  if (tile >= (long)(M >> 4) * ntiles) return;
  const int mt = (int)(tile / ntiles);
  const int nt = (int)(tile - (long)mt * ntiles);

  const int am = mt * 16 + l15;   // A row handled by this lane
  const int bn = nt * 16 + l15;   // B column (row of WT) for this lane

  v8f acc = {};
  for (int k0 = 0; k0 < K; k0 += 32) {
    Frag a, b;
    const int ka = k0 + half * 8;
    if (AF32) {
      const float* ap = (const float*)Ap + (long)am * K + ka;
      a.b4[0] = __builtin_convertvector(*(const v4f*)(ap),      v4bf);
      a.b4[1] = __builtin_convertvector(*(const v4f*)(ap + 4),  v4bf);
      a.b4[2] = __builtin_convertvector(*(const v4f*)(ap + 16), v4bf);
      a.b4[3] = __builtin_convertvector(*(const v4f*)(ap + 20), v4bf);
    } else {
      const u16* ap = (const u16*)Ap + (long)am * K + ka;
      a.q[0] = *(const u32x4*)(ap);
      a.q[1] = *(const u32x4*)(ap + 16);
    }
    const u16* bp = WT + (long)bn * K + k0 + half * 8;
    b.q[0] = *(const u32x4*)(bp);
    b.q[1] = *(const u32x4*)(bp + 16);
    acc = wmma_bf16(a, b, acc);
  }

  const float bcol = bias[bn];
  if (MODE == 0) {
    float* o = (float*)outp;
#pragma unroll
    for (int r = 0; r < 8; ++r) {
      const int m = mt * 16 + r + half * 8;
      o[(long)m * N + bn] = (acc[r] + bcol) * alpha;
    }
  } else {
    u16* o = (u16*)outp;
    const int hh = bn >> 5, dd = bn & 31;
    // a 16-row tile never spans a batch boundary (rows % 16 == 0)
    const int bb  = (mt * 16) / rows;
    const int tt0 = mt * 16 - bb * rows;
#pragma unroll
    for (int r = 0; r < 8; ++r) {
      const int tt = tt0 + r + half * 8;
      const float val = (acc[r] + bcol) * alpha;
      long idx;
      if (MODE == 1)
        idx = (((long)bb * 8 + hh) * rows + tt) * 32 + dd;
      else
        idx = (((long)bb * 8 + hh) * 32 + dd) * (long)rows + tt;
      o[idx] = f2bf(val);
    }
  }
}

// ---------------------------------------------------------------------------
// Weight convert + transpose: W[K][N] f32 -> WT[N][K] bf16
// ---------------------------------------------------------------------------
__global__ __launch_bounds__(256)
void convert_transpose(const float* __restrict__ W, u16* __restrict__ WT,
                       int K, int N)
{
  const long i = (long)blockIdx.x * 256 + threadIdx.x;
  if (i >= (long)K * N) return;
  const int k = (int)(i / N);
  const int n = (int)(i - (long)k * N);
  WT[(long)n * K + k] = f2bf(W[i]);
}

// ---------------------------------------------------------------------------
// attn_v: per wave, 16 t-rows x all 256 s. QK^T (16 WMMAs), masked softmax
// over S with shfl reductions, stage P to LDS in A-layout, P@vl (16 WMMAs),
// store bf16 context [B,T,E].
// grid (T/128, H, B), block 256 (8 waves). dyn LDS: 8*16*272*2 bytes.
// ---------------------------------------------------------------------------
__global__ __launch_bounds__(256)
void attn_v_kernel(const u16* __restrict__ q, const u16* __restrict__ kmat,
                   const u16* __restrict__ vlT, const int* __restrict__ mask,
                   u16* __restrict__ ctx_v)
{
  extern __shared__ u16 smem[];
  const int lane = threadIdx.x & 31;
  const int wave = threadIdx.x >> 5;
  const int half = lane >> 4;
  const int l15  = lane & 15;
  const int b = blockIdx.z, h = blockIdx.y;
  const long bh = (long)b * 8 + h;
  const int tw = blockIdx.x * 128 + wave * 16;

  const u16* qbh  = q    + (bh * 8192 + tw) * 32;
  const u16* kbh  = kmat + bh * 256 * 32;
  const u16* vlbh = vlT  + bh * 32 * 256;
  const int* mk   = mask + b * 256;

  Frag a;
  { const u16* ap = qbh + l15 * 32 + half * 8;
    a.q[0] = *(const u32x4*)ap;
    a.q[1] = *(const u32x4*)(ap + 16); }

  v8f  w[16];
  float biasv[16];
#pragma unroll
  for (int st = 0; st < 16; ++st) {
    Frag kf;
    const u16* bp = kbh + (st * 16 + l15) * 32 + half * 8;
    kf.q[0] = *(const u32x4*)bp;
    kf.q[1] = *(const u32x4*)(bp + 16);
    v8f acc = {};
    acc = wmma_bf16(a, kf, acc);
#pragma unroll
    for (int r = 0; r < 8; ++r)
      acc[r] = fminf(fmaxf(acc[r], -50000.0f), 50000.0f);
    w[st] = acc;
    biasv[st] = (mk[st * 16 + l15] == 0) ? -9.0e15f : 0.0f;
  }

  float rmax[8], rsum[8], rinv[8];
#pragma unroll
  for (int r = 0; r < 8; ++r) rmax[r] = -3.0e38f;
#pragma unroll
  for (int st = 0; st < 16; ++st)
#pragma unroll
    for (int r = 0; r < 8; ++r)
      rmax[r] = fmaxf(rmax[r], w[st][r] + biasv[st]);
#pragma unroll
  for (int r = 0; r < 8; ++r) {
    float m = rmax[r];
    m = fmaxf(m, __shfl_xor(m, 1));
    m = fmaxf(m, __shfl_xor(m, 2));
    m = fmaxf(m, __shfl_xor(m, 4));
    m = fmaxf(m, __shfl_xor(m, 8));
    rmax[r] = m;
  }
#pragma unroll
  for (int r = 0; r < 8; ++r) rsum[r] = 0.0f;
  // exp once: overwrite w[] with un-normalized probabilities
#pragma unroll
  for (int st = 0; st < 16; ++st)
#pragma unroll
    for (int r = 0; r < 8; ++r) {
      const float e = __expf(w[st][r] + biasv[st] - rmax[r]);
      rsum[r] += e;
      w[st][r] = e;
    }
#pragma unroll
  for (int r = 0; r < 8; ++r) {
    float s = rsum[r];
    s += __shfl_xor(s, 1);
    s += __shfl_xor(s, 2);
    s += __shfl_xor(s, 4);
    s += __shfl_xor(s, 8);
    rinv[r] = 1.0f / s;
  }

  // stage normalized P (bf16) to per-wave LDS: [16 t][pitch 272]
  u16* pw = smem + wave * (16 * 272);
#pragma unroll
  for (int st = 0; st < 16; ++st) {
    const int s = st * 16 + l15;
#pragma unroll
    for (int r = 0; r < 8; ++r) {
      const int m = r + half * 8;
      pw[m * 272 + s] = f2bf(w[st][r] * rinv[r]);
    }
  }
  __syncthreads();

  v8f ov0 = {}, ov1 = {};
#pragma unroll
  for (int sp = 0; sp < 8; ++sp) {         // K = 32 s-values per step
    Frag pa;
    const u16* ap2 = pw + l15 * 272 + sp * 32 + half * 8;
    pa.q[0] = *(const u32x4*)ap2;
    pa.q[1] = *(const u32x4*)(ap2 + 16);
    Frag b0, b1;
    const u16* bp0 = vlbh + (long)(l15)      * 256 + sp * 32 + half * 8;
    const u16* bp1 = vlbh + (long)(16 + l15) * 256 + sp * 32 + half * 8;
    b0.q[0] = *(const u32x4*)bp0; b0.q[1] = *(const u32x4*)(bp0 + 16);
    b1.q[0] = *(const u32x4*)bp1; b1.q[1] = *(const u32x4*)(bp1 + 16);
    ov0 = wmma_bf16(pa, b0, ov0);
    ov1 = wmma_bf16(pa, b1, ov1);
  }

  u16* co = ctx_v + (((long)b * 8192 + tw) * 256) + h * 32;
#pragma unroll
  for (int r = 0; r < 8; ++r) {
    const int m = r + half * 8;
    co[(long)m * 256 + l15]      = f2bf(ov0[r]);
    co[(long)m * 256 + 16 + l15] = f2bf(ov1[r]);
  }
}

// ---------------------------------------------------------------------------
// attn_l: block owns one 16-column s-tile for all T. Pass1: col max over T.
// Pass2: P = exp(w - colmax), colsum, and P^T @ vv (LDS transpose stage),
// fixed-order cross-wave reduction -> deterministic, no atomics.
// grid (S/16, H, B), block 256 (8 waves).
// ---------------------------------------------------------------------------
__global__ __launch_bounds__(256)
void attn_l_kernel(const u16* __restrict__ q, const u16* __restrict__ kmat,
                   const u16* __restrict__ vvT, u16* __restrict__ ctx_l)
{
  __shared__ float cmred[8][16];
  __shared__ float csred[8][16];
  __shared__ float numred[8][16][32];
  __shared__ u16  pT[8][16 * 40];   // per-wave [16 s][pitch 40] over 32 t

  const int lane = threadIdx.x & 31;
  const int wave = threadIdx.x >> 5;
  const int half = lane >> 4;
  const int l15  = lane & 15;
  const int b = blockIdx.z, h = blockIdx.y;
  const long bh = (long)b * 8 + h;
  const int sbase = blockIdx.x * 16;

  const u16* qbh  = q    + bh * 8192 * 32;
  const u16* kbh  = kmat + bh * 256 * 32;
  const u16* vvbh = vvT  + bh * 32 * 8192;

  Frag kf;
  { const u16* bp = kbh + (sbase + l15) * 32 + half * 8;
    kf.q[0] = *(const u32x4*)bp;
    kf.q[1] = *(const u32x4*)(bp + 16); }

  // pass 1: column max over all T (recompute QK^T; flops are free here)
  float cmax = -3.0e38f;
  for (int it = 0; it < 32; ++it) {
    const int t0 = (it * 8 + wave) * 32;
#pragma unroll
    for (int sub = 0; sub < 2; ++sub) {
      Frag a;
      const u16* ap = qbh + (long)(t0 + sub * 16 + l15) * 32 + half * 8;
      a.q[0] = *(const u32x4*)ap;
      a.q[1] = *(const u32x4*)(ap + 16);
      v8f wv = {};
      wv = wmma_bf16(a, kf, wv);
#pragma unroll
      for (int r = 0; r < 8; ++r)
        cmax = fmaxf(cmax, fminf(fmaxf(wv[r], -50000.0f), 50000.0f));
    }
  }
  cmax = fmaxf(cmax, __shfl_xor(cmax, 16));
  if (half == 0) cmred[wave][l15] = cmax;
  __syncthreads();
  float cmv = cmred[0][l15];
#pragma unroll
  for (int wv2 = 1; wv2 < 8; ++wv2) cmv = fmaxf(cmv, cmred[wv2][l15]);

  // pass 2: exp / colsum / P^T @ vv
  v8f  on0 = {}, on1 = {};
  float csum = 0.0f;
  u16* myp = pT[wave];
  for (int it = 0; it < 32; ++it) {
    const int t0 = (it * 8 + wave) * 32;
#pragma unroll
    for (int sub = 0; sub < 2; ++sub) {
      Frag a;
      const u16* ap = qbh + (long)(t0 + sub * 16 + l15) * 32 + half * 8;
      a.q[0] = *(const u32x4*)ap;
      a.q[1] = *(const u32x4*)(ap + 16);
      v8f wv = {};
      wv = wmma_bf16(a, kf, wv);
#pragma unroll
      for (int r = 0; r < 8; ++r) {
        const float x = fminf(fmaxf(wv[r], -50000.0f), 50000.0f);
        const float p = __expf(x - cmv);
        csum += p;
        myp[l15 * 40 + sub * 16 + r + half * 8] = f2bf(p);   // transpose: [s][t]
      }
    }
    __syncthreads();
    Frag pa;                       // A = P^T tile: 16 s x 32 t
    const u16* ap2 = myp + l15 * 40 + half * 8;
    pa.q[0] = *(const u32x4*)ap2;
    pa.q[1] = *(const u32x4*)(ap2 + 16);
    Frag b0, b1;
    const u16* bp0 = vvbh + (long)l15        * 8192 + t0 + half * 8;
    const u16* bp1 = vvbh + (long)(16 + l15) * 8192 + t0 + half * 8;
    b0.q[0] = *(const u32x4*)bp0; b0.q[1] = *(const u32x4*)(bp0 + 16);
    b1.q[0] = *(const u32x4*)bp1; b1.q[1] = *(const u32x4*)(bp1 + 16);
    on0 = wmma_bf16(pa, b0, on0);
    on1 = wmma_bf16(pa, b1, on1);
    __syncthreads();
  }

  csum += __shfl_xor(csum, 16);
  if (half == 0) csred[wave][l15] = csum;
#pragma unroll
  for (int r = 0; r < 8; ++r) {
    const int sl = r + half * 8;
    numred[wave][sl][l15]      = on0[r];
    numred[wave][sl][16 + l15] = on1[r];
  }
  __syncthreads();

  for (int e = threadIdx.x; e < 512; e += 256) {
    const int sl = e >> 5, d = e & 31;
    float ns = 0.0f, cs = 0.0f;
#pragma unroll
    for (int wv2 = 0; wv2 < 8; ++wv2) { ns += numred[wv2][sl][d]; cs += csred[wv2][sl]; }
    ctx_l[(((long)b * 256 + sbase + sl) * 256) + h * 32 + d] = f2bf(ns / cs);
  }
}

// ---------------------------------------------------------------------------
// Host orchestration
// ---------------------------------------------------------------------------
extern "C" void kernel_launch(void* const* d_in, const int* in_sizes, int n_in,
                              void* d_out, int out_size, void* d_ws, size_t ws_size,
                              hipStream_t stream)
{
  (void)in_sizes; (void)n_in; (void)out_size; (void)ws_size;
  const float* v    = (const float*)d_in[0];
  const float* l    = (const float*)d_in[1];
  const int*   mask = (const int*)d_in[2];
  const float* Wq   = (const float*)d_in[3];
  const float* bq   = (const float*)d_in[4];
  const float* Wk   = (const float*)d_in[5];
  const float* bk   = (const float*)d_in[6];
  const float* Wvv  = (const float*)d_in[7];
  const float* bvv  = (const float*)d_in[8];
  const float* Wvl  = (const float*)d_in[9];
  const float* bvl  = (const float*)d_in[10];
  const float* Wov  = (const float*)d_in[11];
  const float* bov  = (const float*)d_in[12];
  const float* Wol  = (const float*)d_in[13];
  const float* bol  = (const float*)d_in[14];

  char* ws = (char*)d_ws;
  size_t off = 0;
  auto alloc = [&](size_t bytes) -> void* {
    void* p = ws + off;
    off = (off + bytes + 255) & ~(size_t)255;
    return p;
  };

  u16* WqT  = (u16*)alloc((size_t)256 * 256 * 2);
  u16* WkT  = (u16*)alloc((size_t)256 * 768 * 2);
  u16* WvvT = (u16*)alloc((size_t)256 * 256 * 2);
  u16* WvlT = (u16*)alloc((size_t)256 * 768 * 2);
  u16* WovT = (u16*)alloc((size_t)256 * 256 * 2);
  u16* WolT = (u16*)alloc((size_t)768 * 256 * 2);
  u16* qb   = (u16*)alloc((size_t)8 * 8 * 8192 * 32 * 2);  // [B,H,T,D] bf16
  u16* kb   = (u16*)alloc((size_t)8 * 8 * 256 * 32 * 2);   // [B,H,S,D]
  u16* vlT  = (u16*)alloc((size_t)8 * 8 * 32 * 256 * 2);   // [B,H,D,S]
  u16* vvT  = (u16*)alloc((size_t)8 * 8 * 32 * 8192 * 2);  // [B,H,D,T]
  u16* ctxv = (u16*)alloc((size_t)8 * 8192 * 256 * 2);     // [B,T,E]
  u16* ctxl = (u16*)alloc((size_t)8 * 256 * 256 * 2);      // [B,S,E]

  convert_transpose<<<(256*256 + 255) / 256, 256, 0, stream>>>(Wq,  WqT,  256, 256);
  convert_transpose<<<(768*256 + 255) / 256, 256, 0, stream>>>(Wk,  WkT,  768, 256);
  convert_transpose<<<(256*256 + 255) / 256, 256, 0, stream>>>(Wvv, WvvT, 256, 256);
  convert_transpose<<<(768*256 + 255) / 256, 256, 0, stream>>>(Wvl, WvlT, 768, 256);
  convert_transpose<<<(256*256 + 255) / 256, 256, 0, stream>>>(Wov, WovT, 256, 256);
  convert_transpose<<<(256*768 + 255) / 256, 256, 0, stream>>>(Wol, WolT, 256, 768);

  const float SCALE = 0.1767766952966369f;  // 32^-0.5
  // projections (f32 A, bf16 head-layout outputs)
  gemm_wmma<true, 1><<<8192, 256, 0, stream>>>(v, WqT,  bq,  SCALE, qb,  65536, 256, 256, 8192);
  gemm_wmma<true, 2><<<8192, 256, 0, stream>>>(v, WvvT, bvv, 1.0f,  vvT, 65536, 256, 256, 8192);
  gemm_wmma<true, 1><<< 256, 256, 0, stream>>>(l, WkT,  bk,  1.0f,  kb,   2048, 256, 768,  256);
  gemm_wmma<true, 2><<< 256, 256, 0, stream>>>(l, WvlT, bvl, 1.0f,  vlT,  2048, 256, 768,  256);

  // fused bidirectional attention
  dim3 gv(64, 8, 8);
  attn_v_kernel<<<gv, 256, (size_t)8 * 16 * 272 * sizeof(u16), stream>>>(qb, kb, vlT, mask, ctxv);
  dim3 gl(16, 8, 8);
  attn_l_kernel<<<gl, 256, 0, stream>>>(qb, kb, vvT, ctxl);

  // output projections (bf16 A, f32 outputs, concatenated out_v then out_l)
  float* outv = (float*)d_out;
  float* outl = outv + (size_t)8 * 8192 * 256;
  gemm_wmma<false, 0><<<8192, 256, 0, stream>>>(ctxv, WovT, bov, 1.0f, outv, 65536, 256, 256, 1);
  gemm_wmma<false, 0><<< 768, 256, 0, stream>>>(ctxl, WolT, bol, 1.0f, outl,  2048, 768, 256, 1);
}
